// FnnMlpMlpRdm_13795434955131
// MI455X (gfx1250) — matrix operand order repaired and checked by reading, not compiled, not run on hardware
//
#include <hip/hip_runtime.h>
#include <cstdint>

typedef __bf16 bf16;
typedef __attribute__((ext_vector_type(16))) __bf16 v16bf;
typedef __attribute__((ext_vector_type(2)))  __bf16 bf16x2;
typedef __attribute__((ext_vector_type(8)))  float   v8f;
typedef __attribute__((ext_vector_type(2)))  float   f32x2;
typedef __attribute__((ext_vector_type(4)))  unsigned v4u;
typedef __attribute__((ext_vector_type(8)))  int      v8i;
typedef __attribute__((ext_vector_type(4)))  int      v4i;

constexpr int kB  = 16384;
constexpr int kF  = 256;
constexpr int kR  = 32;
constexpr int kC  = 16;
constexpr int kH1 = 128;
constexpr int kH2 = 32;
constexpr int kG1 = 256;
constexpr int kG2 = 64;

// conseq_kernel LDS arena layout (dynamic shared, base offset 0):
//   [0, 131072)           : f32 staging tile [128][256]  (TDM destination)
//                           ... later aliased by c1T bf16 [128][kG1+8]
//   [131072, 198656)      : dataT bf16 [128][kF+8]
//   [198656, 217088)      : c2T   bf16 [128][kG2+8]
constexpr int kStageBytes = 128 * kF * 4;                    // 131072
constexpr int kDataTOff   = kStageBytes;                     // 131072
constexpr int kC2TOff     = kDataTOff + 128 * (kF + 8) * 2;  // 198656
constexpr int kSmemC      = kC2TOff + 128 * (kG2 + 8) * 2;   // 217088

__device__ __forceinline__ v8f wmma_bf16(v16bf a, v16bf b, v8f c) {
  return __builtin_amdgcn_wmma_f32_16x16x32_bf16(false, a, false, b, (short)0, c,
                                                 false, false);
}

// A fragment (16x32 bf16, M x K) from a row-major LDS tile, leading dim ld.
__device__ __forceinline__ v16bf load_a_frag(const bf16* tile, int ld, int m0,
                                             int k0, int lane) {
  const int half = (lane >> 4) & 1;
  const bf16* row = tile + (size_t)(m0 + (lane & 15)) * ld;
  v16bf a;
#pragma unroll
  for (int j = 0; j < 8; ++j) {
    const int kb = k0 + ((j < 4) ? (2 * j) : (16 + 2 * (j - 4))) + half * 8;
    bf16x2 p = *(const bf16x2*)(row + kb);
    a[2 * j]     = p.x;
    a[2 * j + 1] = p.y;
  }
  return a;
}

// B fragment (32x16, K x N) from a row-major [out][in] f32 weight matrix.
__device__ __forceinline__ v16bf load_b_frag(const float* W, int ld, int k0,
                                             int n0, int lane) {
  const float* wr = W + (size_t)(n0 + (lane & 15)) * ld + k0 + ((lane >> 4) << 3);
  v16bf b;
#pragma unroll
  for (int j = 0; j < 8; ++j) {
    f32x2 w = *(const f32x2*)(wr + ((j < 4) ? (2 * j) : (16 + 2 * (j - 4))));
    b[2 * j]     = (bf16)w.x;
    b[2 * j + 1] = (bf16)w.y;
  }
  return b;
}

// ---------------------------------------------------------------------------
// Kernel 1: fused Gaussian membership + firing-strength MLP.
// ---------------------------------------------------------------------------
__global__ __launch_bounds__(256) void fs_kernel(
    const float* __restrict__ data, const float* __restrict__ proto,
    const float* __restrict__ var,  const float* __restrict__ Wf1,
    const float* __restrict__ bf1,  const float* __restrict__ Wf2,
    const float* __restrict__ bf2,  const float* __restrict__ Wf3,
    const float* __restrict__ bf3,  float* __restrict__ fs_ini) {
  __shared__ alignas(16) bf16 memT[128][kF + 8];
  __shared__ alignas(16) bf16 h1T[128][kH1 + 8];
  __shared__ alignas(16) bf16 h2T[128][kH2 + 8];

  const int r    = blockIdx.y;
  const int b0   = blockIdx.x * 128;
  const int t    = threadIdx.x;
  const int lane = t & 31;
  const int wave = t >> 5;

  // Phase 1: membership tile -> LDS bf16 (never touches HBM)
  for (int idx = t; idx < 128 * kF; idx += 256) {
    const int row = idx >> 8;
    const int f   = idx & (kF - 1);
    const float d = data[(size_t)(b0 + row) * kF + f];
    const float p = proto[r * kF + f];
    float v = var[r * kF + f];
    v = fminf(fmaxf(v, 1e-4f), 0.1f);
    const float x = d - p;
    memT[row][f] = (bf16)__expf(-(x * x) / (2.0f * v * v));
  }
  __syncthreads();

  // Phase 2: h1 = relu(mem @ Wf1^T + bf1): wave w owns N-tile w, loops 8 M-tiles
  {
    const int n0 = wave * 16;
    v16bf bw[8];
#pragma unroll
    for (int kk = 0; kk < 8; ++kk)
      bw[kk] = load_b_frag(Wf1, kF, kk * 32, n0, lane);
    const int n  = n0 + (lane & 15);
    const float bias = bf1[n];
    for (int mt = 0; mt < 8; ++mt) {
      const int m0 = mt * 16;
      v8f acc = {};
#pragma unroll
      for (int kk = 0; kk < 8; ++kk) {
        v16bf a = load_a_frag(&memT[0][0], kF + 8, m0, kk * 32, lane);
        acc = wmma_bf16(a, bw[kk], acc);
      }
      const int mb = m0 + ((lane >> 4) << 3);
#pragma unroll
      for (int i = 0; i < 8; ++i)
        h1T[mb + i][n] = (bf16)fmaxf(acc[i] + bias, 0.0f);
    }
  }
  __syncthreads();

  // Phase 3: h2 = relu(h1 @ Wf2^T + bf2): wave w owns N-tile (w&1),
  // M-tiles (w>>1) and (w>>1)+4.  B fragments invariant per wave.
  {
    const int n0 = (wave & 1) * 16;
    v16bf bw[4];
#pragma unroll
    for (int kk = 0; kk < 4; ++kk)
      bw[kk] = load_b_frag(Wf2, kH1, kk * 32, n0, lane);
    const int n  = n0 + (lane & 15);
    const float bias = bf2[n];
#pragma unroll
    for (int half = 0; half < 2; ++half) {
      const int m0 = ((wave >> 1) + half * 4) * 16;
      v8f acc = {};
#pragma unroll
      for (int kk = 0; kk < 4; ++kk) {
        v16bf a = load_a_frag(&h1T[0][0], kH1 + 8, m0, kk * 32, lane);
        acc = wmma_bf16(a, bw[kk], acc);
      }
      const int mb = m0 + ((lane >> 4) << 3);
#pragma unroll
      for (int i = 0; i < 8; ++i)
        h2T[mb + i][n] = (bf16)fmaxf(acc[i] + bias, 0.0f);
    }
  }
  __syncthreads();

  // Phase 4: fs_ini = relu(h2 . Wf3 + bf3)
  if (t < 128) {
    float s = bf3[0];
#pragma unroll
    for (int p = 0; p < kH2; ++p)
      s += (float)h2T[t][p] * Wf3[p];
    fs_ini[(size_t)r * kB + (b0 + t)] = fmaxf(s, 0.0f);
  }
}

// ---------------------------------------------------------------------------
// Kernel 2: softmax over rules -> fire_strength [B][R]
// ---------------------------------------------------------------------------
__global__ __launch_bounds__(256) void softmax_kernel(
    const float* __restrict__ fs_ini, float* __restrict__ fire) {
  const int b = blockIdx.x * 256 + threadIdx.x;
  float v[kR];
  float mx = -3.4e38f;
#pragma unroll
  for (int r = 0; r < kR; ++r) {
    v[r] = fs_ini[(size_t)r * kB + b];
    mx = fmaxf(mx, v[r]);
  }
  float sum = 0.0f;
#pragma unroll
  for (int r = 0; r < kR; ++r) {
    v[r] = __expf(v[r] - mx);
    sum += v[r];
  }
  const float inv = 1.0f / sum;
#pragma unroll
  for (int r = 0; r < kR; ++r)
    fire[(size_t)b * kR + r] = v[r] * inv;
}

// ---------------------------------------------------------------------------
// Kernel 3: 32 per-rule consequent MLPs + softmax-weighted reduction.
// Data tile staged into LDS by the Tensor Data Mover, then converted to bf16.
// ---------------------------------------------------------------------------
__global__ __launch_bounds__(256) void conseq_kernel(
    const float* __restrict__ data,
    const float* __restrict__ Wc1, const float* __restrict__ bc1,
    const float* __restrict__ Wc2, const float* __restrict__ bc2,
    const float* __restrict__ Wc3, const float* __restrict__ bc3,
    const float* __restrict__ fire, float* __restrict__ out) {
  extern __shared__ char smem[];
  float* stage = (float*)smem;                       // [128][kF] f32 (offset 0)
  bf16* c1T    = (bf16*)smem;                        // [128][kG1+8], aliases stage
  bf16* dataT  = (bf16*)(smem + kDataTOff);          // [128][kF+8]
  bf16* c2T    = (bf16*)(smem + kC2TOff);            // [128][kG2+8]
  const int ld1 = kG1 + 8, ldD = kF + 8, ld2 = kG2 + 8;

  const int b0   = blockIdx.x * 128;
  const int t    = threadIdx.x;
  const int lane = t & 31;
  const int wave = t >> 5;

#if __has_builtin(__builtin_amdgcn_tensor_load_to_lds) && \
    __has_builtin(__builtin_amdgcn_s_wait_tensorcnt)
  // --- TDM: DMA the 128x256 f32 data tile into LDS offset 0 -----------------
  if (wave == 0) {
    const unsigned long long ga =
        (unsigned long long)(uintptr_t)(data + (size_t)b0 * kF);
    v4u g0;
    g0[0] = 1u;                                   // count=1, user descriptor
    g0[1] = 0u;                                   // lds_addr = 0 (stage)
    g0[2] = (unsigned)(ga & 0xFFFFFFFFull);       // global_addr[31:0]
    g0[3] = (unsigned)((ga >> 32) & 0x1FFFFFFull) // global_addr[56:32]
            | (2u << 30);                         // type = 2 ("image")
    v8i g1;
    g1[0] = 0x00020000;                 // wg_mask=0, data_size=2 (4 bytes)
    g1[1] = (kF & 0xFFFF) << 16;        // tensor_dim0[15:0]=256
    g1[2] = (kF >> 16) | (128 << 16);   // tensor_dim0[31:16] | tensor_dim1=128
    g1[3] = (kF << 16);                 // tensor_dim1[31:16]=0 | tile_dim0=256
    g1[4] = 128;                        // tile_dim1=128, tile_dim2=0
    g1[5] = kF;                         // tensor_dim0_stride = 256 (low 32)
    g1[6] = 0;
    g1[7] = 0;
    v4i gz4 = {0, 0, 0, 0};                       // 2D tensor: groups 2/3 unused
    v8i gz8 = {0, 0, 0, 0, 0, 0, 0, 0};
    __builtin_amdgcn_tensor_load_to_lds(g0, g1, gz4, gz4, gz8, 0);
    __builtin_amdgcn_s_wait_tensorcnt(0);
  }
  __syncthreads();
#else
  for (int idx = t; idx < 128 * kF; idx += 256)
    stage[idx] = data[(size_t)b0 * kF + idx];
  __syncthreads();
#endif

  // Convert f32 staging tile -> bf16 dataT (stage is dead afterwards; its
  // space is reused as c1T).
  for (int idx = t; idx < 128 * kF; idx += 256) {
    const int row = idx >> 8;
    const int f   = idx & (kF - 1);
    dataT[row * ldD + f] = (bf16)stage[idx];
  }
  __syncthreads();

  const int m0 = wave * 16;                 // this wave's 16 output rows
  const int n  = lane & 15;                 // output column (C == 16)
  const int mb = m0 + ((lane >> 4) << 3);
  v8f outacc = {};

  for (int r = 0; r < kR; ++r) {
    // Prefetch next rule's weights into cache while this rule computes.
    if (r + 1 < kR) {
      const char* nw1 = (const char*)(Wc1 + (size_t)(r + 1) * kG1 * kF);
#pragma unroll
      for (int pf = 0; pf < 8; ++pf)
        __builtin_prefetch(nw1 + (t + pf * 256) * 128, 0, 1);
      const char* nw2 = (const char*)(Wc2 + (size_t)(r + 1) * kG2 * kG1);
#pragma unroll
      for (int pf = 0; pf < 2; ++pf)
        __builtin_prefetch(nw2 + (t + pf * 256) * 128, 0, 1);
    }

    // c1 = elu(data @ Wc1[r]^T + bc1[r]) : [128][256]
    // Wave w owns N-tiles {2w, 2w+1}; B fragments cached across 8 M-tiles.
    const float* W1 = Wc1 + (size_t)r * kG1 * kF;
    const float* B1 = bc1 + r * kG1;
#pragma unroll
    for (int cidx = 0; cidx < 2; ++cidx) {
      const int tn0 = (wave * 2 + cidx) * 16;
      v16bf bw[8];
#pragma unroll
      for (int kk = 0; kk < 8; ++kk)
        bw[kk] = load_b_frag(W1, kF, kk * 32, tn0, lane);
      const int tn = tn0 + (lane & 15);
      const float bias = B1[tn];
      for (int mt = 0; mt < 8; ++mt) {
        const int tm0 = mt * 16;
        v8f acc = {};
#pragma unroll
        for (int kk = 0; kk < 8; ++kk) {
          v16bf a = load_a_frag(dataT, ldD, tm0, kk * 32, lane);
          acc = wmma_bf16(a, bw[kk], acc);
        }
        const int tmb = tm0 + ((lane >> 4) << 3);
#pragma unroll
        for (int i = 0; i < 8; ++i) {
          const float x = acc[i] + bias;
          c1T[(tmb + i) * ld1 + tn] = (bf16)(x > 0.0f ? x : __expf(x) - 1.0f);
        }
      }
    }
    __syncthreads();

    // c2 = elu(c1 @ Wc2[r]^T + bc2[r]) : [128][64]
    // Wave w owns N-tile (w&3); M-tiles (w>>2)*4 .. +3.  B cached.
    const float* W2 = Wc2 + (size_t)r * kG2 * kG1;
    const float* B2 = bc2 + r * kG2;
    {
      const int tn0 = (wave & 3) * 16;
      v16bf bw[8];
#pragma unroll
      for (int kk = 0; kk < 8; ++kk)
        bw[kk] = load_b_frag(W2, kG1, kk * 32, tn0, lane);
      const int tn = tn0 + (lane & 15);
      const float bias = B2[tn];
#pragma unroll
      for (int mt4 = 0; mt4 < 4; ++mt4) {
        const int tm0 = ((wave >> 2) * 4 + mt4) * 16;
        v8f acc = {};
#pragma unroll
        for (int kk = 0; kk < 8; ++kk) {
          v16bf a = load_a_frag(c1T, ld1, tm0, kk * 32, lane);
          acc = wmma_bf16(a, bw[kk], acc);
        }
        const int tmb = tm0 + ((lane >> 4) << 3);
#pragma unroll
        for (int i = 0; i < 8; ++i) {
          const float x = acc[i] + bias;
          c2T[(tmb + i) * ld2 + tn] = (bf16)(x > 0.0f ? x : __expf(x) - 1.0f);
        }
      }
    }
    __syncthreads();

    // c3 = relu(c2 @ Wc3[r]^T + bc3[r]) : one 16x16 tile per wave, then
    // fire-strength-weighted accumulation into registers.
    const float* W3 = Wc3 + (size_t)r * kC * kG2;
    const float* B3 = bc3 + r * kC;
    v8f acc = {};
#pragma unroll
    for (int k0 = 0; k0 < kG2; k0 += 32) {
      v16bf a = load_a_frag(c2T, ld2, m0, k0, lane);
      v16bf b = load_b_frag(W3, kG2, k0, 0, lane);
      acc = wmma_bf16(a, b, acc);
    }
    const float bias = B3[n];
#pragma unroll
    for (int i = 0; i < 8; ++i) {
      const float c3v = fmaxf(acc[i] + bias, 0.0f);
      const float fsv = fire[(size_t)(b0 + mb + i) * kR + r];
      outacc[i] += c3v * fsv;
    }
    __syncthreads();   // c1T/c2T free for next rule
  }

#pragma unroll
  for (int i = 0; i < 8; ++i)
    out[(size_t)(b0 + mb + i) * kC + n] = outacc[i];
}

// ---------------------------------------------------------------------------
extern "C" void kernel_launch(void* const* d_in, const int* in_sizes, int n_in,
                              void* d_out, int out_size, void* d_ws,
                              size_t ws_size, hipStream_t stream) {
  const float* data  = (const float*)d_in[0];
  const float* proto = (const float*)d_in[1];
  const float* var   = (const float*)d_in[2];
  const float* Wf1   = (const float*)d_in[3];
  const float* bf1   = (const float*)d_in[4];
  const float* Wf2   = (const float*)d_in[5];
  const float* bf2   = (const float*)d_in[6];
  const float* Wf3   = (const float*)d_in[7];
  const float* bf3   = (const float*)d_in[8];
  const float* Wc1   = (const float*)d_in[9];
  const float* bc1   = (const float*)d_in[10];
  const float* Wc2   = (const float*)d_in[11];
  const float* bc2   = (const float*)d_in[12];
  const float* Wc3   = (const float*)d_in[13];
  const float* bc3   = (const float*)d_in[14];

  float* out    = (float*)d_out;              // outputs [B, C]
  float* fire   = out + (size_t)kB * kC;      // fire_strength [B, R]
  float* fs_ini = (float*)d_ws;               // scratch [R, B] (2 MB)

  dim3 gridA(kB / 128, kR);
  fs_kernel<<<gridA, 256, 0, stream>>>(data, proto, var, Wf1, bf1, Wf2, bf2,
                                       Wf3, bf3, fs_ini);
  softmax_kernel<<<kB / 256, 256, 0, stream>>>(fs_ini, fire);
  conseq_kernel<<<kB / 128, 256, kSmemC, stream>>>(data, Wc1, bc1, Wc2, bc2,
                                                   Wc3, bc3, fire, out);
}
